// ContentExtracctor_45835890983468
// MI455X (gfx1250) — matrix operand order repaired and checked by reading, not compiled, not run on hardware
//
#include <hip/hip_runtime.h>

#define D_DIM 768
#define T_DIM 8192
#define N_DIM 16384
#define C_DIM 96
#define NSPLIT 8
#define NT_PER_WG (N_DIM / NSPLIT / 16)   // 128 16-col tiles per workgroup

typedef unsigned short ushort_t;
typedef __attribute__((ext_vector_type(16))) __bf16 v16bf;
typedef __attribute__((ext_vector_type(8)))  __bf16 v8bf;
typedef __attribute__((ext_vector_type(8)))  float  v8f;
typedef __attribute__((ext_vector_type(4)))  unsigned int uint4v;
typedef __attribute__((ext_vector_type(4)))  int int4v;
typedef __attribute__((ext_vector_type(4)))  int v4i;

#define AS1 __attribute__((address_space(1)))
#define AS3 __attribute__((address_space(3)))

#if defined(__HIP_DEVICE_COMPILE__) && \
    __has_builtin(__builtin_amdgcn_global_load_async_to_lds_b128)
#define HAVE_ASYNC_LDS 1
#else
#define HAVE_ASYNC_LDS 0
#endif

#if HAVE_ASYNC_LDS && __has_builtin(__builtin_amdgcn_s_wait_asynccnt)
#define ASYNC_WAIT(n) __builtin_amdgcn_s_wait_asynccnt(n)
#elif HAVE_ASYNC_LDS
#define ASYNC_WAIT(n) asm volatile("s_wait_asynccnt %0" ::"i"(n) : "memory")
#else
#define ASYNC_WAIT(n) ((void)0)
#endif

#if defined(__HIP_DEVICE_COMPILE__) && \
    __has_builtin(__builtin_amdgcn_sched_barrier)
#define SCHED_FENCE() __builtin_amdgcn_sched_barrier(0)
#else
#define SCHED_FENCE() ((void)0)
#endif

__device__ __forceinline__ void copy16_to_lds(const ushort_t* g, ushort_t* l) {
#if HAVE_ASYNC_LDS
    __builtin_amdgcn_global_load_async_to_lds_b128((AS1 v4i*)g, (AS3 v4i*)l,
                                                   0, 0);
#else
    *(uint4v*)l = *(const uint4v*)g;
#endif
}

__device__ __forceinline__ ushort_t f32_to_bf16(float f) {
    unsigned int u = __float_as_uint(f);
    u += 0x7FFFu + ((u >> 16) & 1u);   // round-to-nearest-even
    return (ushort_t)(u >> 16);
}

__device__ __forceinline__ void topk_insert(float v, int n,
                                            float& s0, float& s1, float& s2, float& s3,
                                            int& i0, int& i1, int& i2, int& i3) {
    if (v > s3) {
        if (v > s0)      { s3=s2;i3=i2; s2=s1;i2=i1; s1=s0;i1=i0; s0=v;i0=n; }
        else if (v > s1) { s3=s2;i3=i2; s2=s1;i2=i1; s1=v;i1=n; }
        else if (v > s2) { s3=s2;i3=i2; s2=v;i2=n; }
        else             { s3=v;i3=n; }
    }
}

// ---------------------------------------------------------------- col norms
__global__ __launch_bounds__(256) void colnorm_kernel(
    const float* __restrict__ src, float* __restrict__ inv, int cols) {
    int c = blockIdx.x * 256 + threadIdx.x;
    if (c >= cols) return;
    float s = 0.f;
    for (int d = 0; d < D_DIM; ++d) {
        float v = src[(size_t)d * cols + c];
        s = fmaf(v, v, s);
    }
    inv[c] = rsqrtf(fmaxf(s, 1e-30f));
}

// ------------------------------------------ transpose + normalize + bf16
// src: f32 [D, cols]  ->  dst: bf16 [cols, D], row c scaled by inv[c]
__global__ __launch_bounds__(256) void xpose_norm_kernel(
    const float* __restrict__ src, const float* __restrict__ inv,
    ushort_t* __restrict__ dst, int cols) {
    __shared__ float tile[32][33];
    const int c0 = blockIdx.x * 32;
    const int d0 = blockIdx.y * 32;
    for (int i = threadIdx.y; i < 32; i += 8)
        tile[i][threadIdx.x] = src[(size_t)(d0 + i) * cols + (c0 + threadIdx.x)];
    __syncthreads();
    for (int i = threadIdx.y; i < 32; i += 8) {
        const int c = c0 + i;
        float v = tile[threadIdx.x][i] * inv[c];
        dst[(size_t)c * D_DIM + d0 + threadIdx.x] = f32_to_bf16(v);
    }
}

// ---------------------------------------------------- P = proj_w @ lut (f32)
__global__ __launch_bounds__(256) void pmat_kernel(
    const float* __restrict__ lut, const float* __restrict__ w,
    float* __restrict__ P) {
    const int n  = blockIdx.x * 256 + threadIdx.x;
    const int c0 = blockIdx.y * 8;
    float acc[8] = {0.f, 0.f, 0.f, 0.f, 0.f, 0.f, 0.f, 0.f};
    for (int d = 0; d < D_DIM; ++d) {
        float lv = lut[(size_t)d * N_DIM + n];
#pragma unroll
        for (int j = 0; j < 8; ++j)
            acc[j] = fmaf(w[(size_t)(c0 + j) * D_DIM + d], lv, acc[j]);
    }
#pragma unroll
    for (int j = 0; j < 8; ++j)
        P[(size_t)(c0 + j) * N_DIM + n] = acc[j];
}

// --------------------------------------- fused cosine GEMM + top-4 (WMMA)
// Grid: (T/128, NSPLIT). Block: 256 (8 waves, 16 T-rows each).
// B panels double-buffered in LDS via async global->LDS copies.
__global__ __launch_bounds__(256, 1) void cos_topk_kernel(
    const ushort_t* __restrict__ xb, const ushort_t* __restrict__ lb,
    float* __restrict__ pval, int* __restrict__ pidx) {
    __shared__ __align__(32) ushort_t Bbuf[2][16 * D_DIM];   // 48 KB
    __shared__ float Stile[8][16][17];                       // 8.5 KB

    const int tid  = threadIdx.x;
    const int wave = tid >> 5;
    const int lane = tid & 31;
    const int t0   = blockIdx.x * 128;
    const int n0   = blockIdx.y * (N_DIM / NSPLIT);

    // Persistent A fragments (contiguous k order, same as the B reads):
    // lanes 0-15: row M=lane,    K = kt*32 + {0..15}
    // lanes 16-31: row M=lane-16, K = kt*32 + {16..31}
    const int am    = lane & 15;
    const int akoff = (lane < 16) ? 0 : 16;
    const ushort_t* arow =
        xb + (size_t)(t0 + wave * 16 + am) * D_DIM + akoff;
    v16bf afrag[24];
#pragma unroll
    for (int kt = 0; kt < 24; ++kt) {
        v8bf lo = *(const v8bf*)(arow + kt * 32);
        v8bf hi = *(const v8bf*)(arow + kt * 32 + 8);
        afrag[kt] = __builtin_shufflevector(lo, hi,
            0, 1, 2, 3, 4, 5, 6, 7, 8, 9, 10, 11, 12, 13, 14, 15);
    }

    const ushort_t* bpanel = lb + (size_t)n0 * D_DIM;   // this split's panel

    // Stage tile 0 into buffer 0 (6 x 16B chunks per thread, contiguous).
#pragma unroll
    for (int j = 0; j < 6; ++j) {
        int off = (tid + j * 256) * 8;
        copy16_to_lds(bpanel + off, &Bbuf[0][off]);
    }

    float s0 = -3.402823466e38f, s1 = s0, s2 = s0, s3 = s0;
    int   i0 = 0, i1 = 0, i2 = 0, i3 = 0;

    const int bcol  = lane & 15;
    const int bkoff = (lane < 16) ? 0 : 16;
    const int srow  = lane & 15;
    const int scol0 = (lane >> 4) * 8;
    const int rbase = (lane >> 4) * 8;

    for (int it = 0; it < NT_PER_WG; ++it) {
        // Prefetch next tile into the other buffer (wraps harmlessly).
        const int nxt = (it + 1) & (NT_PER_WG - 1);
        const ushort_t* nsrc = bpanel + (size_t)nxt * 16 * D_DIM;
        ushort_t* ndst = Bbuf[(it + 1) & 1];
#pragma unroll
        for (int j = 0; j < 6; ++j) {
            int off = (tid + j * 256) * 8;
            copy16_to_lds(nsrc + off, ndst + off);
        }
        ASYNC_WAIT(6);        // this wave's copies for tile `it` are done
        __syncthreads();      // all waves' copies done; prior reads fenced

        const ushort_t* bt = Bbuf[it & 1] + (size_t)bcol * D_DIM + bkoff;
        v8f acc = {0.f, 0.f, 0.f, 0.f, 0.f, 0.f, 0.f, 0.f};
        // Software-pipelined B-fragment reads, 2 fragments in flight. The
        // sched_barrier pins each look-ahead ds_load BEFORE the WMMA that
        // consumes the older fragment, so RA must keep 3 rotating buffers and
        // the per-WMMA wait becomes dscnt<=4 instead of dscnt==0.
        v16bf bb0 = *(const v16bf*)(bt);
        v16bf bb1 = *(const v16bf*)(bt + 32);
#pragma unroll
        for (int kt = 0; kt < 24; ++kt) {
            v16bf bb2 = (kt + 2 < 24) ? *(const v16bf*)(bt + (kt + 2) * 32)
                                      : bb0;
            SCHED_FENCE();
            acc = __builtin_amdgcn_wmma_f32_16x16x32_bf16(
                false, afrag[kt], false, bb0, (short)0, acc, false, false);
            bb0 = bb1;
            bb1 = bb2;
        }

        // Bounce 16x16 f32 tile through per-wave LDS so each lane scans a row.
#pragma unroll
        for (int r = 0; r < 8; ++r)
            Stile[wave][rbase + r][bcol] = acc[r];
        __builtin_amdgcn_wave_barrier();
#pragma unroll
        for (int c = 0; c < 8; ++c) {
            float v = Stile[wave][srow][scol0 + c];
            int   n = n0 + it * 16 + scol0 + c;
            topk_insert(v, n, s0, s1, s2, s3, i0, i1, i2, i3);
        }
        __syncthreads();      // reads of Bbuf[it&1] done before it is re-staged
    }

    // Merge lane m with lane m+16 (they covered the two column halves).
    float o0 = __shfl(s0, lane ^ 16), o1 = __shfl(s1, lane ^ 16);
    float o2 = __shfl(s2, lane ^ 16), o3 = __shfl(s3, lane ^ 16);
    int   q0 = __shfl(i0, lane ^ 16), q1 = __shfl(i1, lane ^ 16);
    int   q2 = __shfl(i2, lane ^ 16), q3 = __shfl(i3, lane ^ 16);
    if (lane < 16) {
        topk_insert(o0, q0, s0, s1, s2, s3, i0, i1, i2, i3);
        topk_insert(o1, q1, s0, s1, s2, s3, i0, i1, i2, i3);
        topk_insert(o2, q2, s0, s1, s2, s3, i0, i1, i2, i3);
        topk_insert(o3, q3, s0, s1, s2, s3, i0, i1, i2, i3);
        const int row = t0 + wave * 16 + lane;
        const size_t base = ((size_t)row * NSPLIT + blockIdx.y) * 4;
        pval[base + 0] = s0; pval[base + 1] = s1;
        pval[base + 2] = s2; pval[base + 3] = s3;
        pidx[base + 0] = i0; pidx[base + 1] = i1;
        pidx[base + 2] = i2; pidx[base + 3] = i3;
    }
}

// ------------------------------------- merge NSPLIT partial top-4 per row
__global__ __launch_bounds__(256) void topk_merge_kernel(
    const float* __restrict__ pval, const int* __restrict__ pidx,
    int* __restrict__ idx_out) {
    const int row = blockIdx.x * 256 + threadIdx.x;   // row < T
    float g0 = -3.402823466e38f, g1 = g0, g2 = g0, g3 = g0;
    int   j0 = 0, j1 = 0, j2 = 0, j3 = 0;
    const float* pv = pval + (size_t)row * NSPLIT * 4;
    const int*   pi = pidx + (size_t)row * NSPLIT * 4;
#pragma unroll
    for (int s = 0; s < NSPLIT * 4; ++s)
        topk_insert(pv[s], pi[s], g0, g1, g2, g3, j0, j1, j2, j3);
    int* o = idx_out + (size_t)row * 4;
    o[0] = j0; o[1] = j1; o[2] = j2; o[3] = j3;
}

// ---------------------------------------------------- gather + bias
__global__ __launch_bounds__(256) void gather_kernel(
    const float* __restrict__ P, const float* __restrict__ bias,
    const int* __restrict__ idx, float* __restrict__ out) {
    const int i = blockIdx.x * 256 + threadIdx.x;   // i < C*T
    const int c = i >> 13;                          // / 8192
    const int t = i & (T_DIM - 1);
    int4v id = *(const int4v*)(idx + (size_t)t * 4);
    const float* Pc = P + (size_t)c * N_DIM;
    float s = Pc[id[0]] + Pc[id[1]] + Pc[id[2]] + Pc[id[3]];
    out[i] = 0.25f * s + bias[c];
}

extern "C" void kernel_launch(void* const* d_in, const int* in_sizes, int n_in,
                              void* d_out, int out_size, void* d_ws, size_t ws_size,
                              hipStream_t stream) {
    const float* x   = (const float*)d_in[0];   // [D, T]
    const float* lut = (const float*)d_in[1];   // [D, N]
    const float* pw  = (const float*)d_in[2];   // [C, D]
    const float* pb  = (const float*)d_in[3];   // [C]
    (void)in_sizes; (void)n_in; (void)out_size; (void)ws_size;

    char* ws = (char*)d_ws;
    ushort_t* xbN  = (ushort_t*)(ws);                    // 12,582,912 B
    ushort_t* lbN  = (ushort_t*)(ws + 12582912);         // 25,165,824 B
    float*    P    = (float*)   (ws + 37748736);         //  6,291,456 B
    int*      idx  = (int*)     (ws + 44040192);         //    131,072 B
    float*    invx = (float*)   (ws + 44171264);         //     32,768 B
    float*    invr = (float*)   (ws + 44204032);         //     65,536 B
    float*    pval = (float*)   (ws + 44269568);         //  1,048,576 B
    int*      pidx = (int*)     (ws + 45318144);         //  1,048,576 B
    // total: 46,366,720 B

    colnorm_kernel<<<T_DIM / 256, 256, 0, stream>>>(x, invx, T_DIM);
    colnorm_kernel<<<N_DIM / 256, 256, 0, stream>>>(lut, invr, N_DIM);

    xpose_norm_kernel<<<dim3(T_DIM / 32, D_DIM / 32), dim3(32, 8), 0, stream>>>(
        x, invx, xbN, T_DIM);
    xpose_norm_kernel<<<dim3(N_DIM / 32, D_DIM / 32), dim3(32, 8), 0, stream>>>(
        lut, invr, lbN, N_DIM);

    pmat_kernel<<<dim3(N_DIM / 256, C_DIM / 8), 256, 0, stream>>>(lut, pw, P);

    cos_topk_kernel<<<dim3(T_DIM / 128, NSPLIT), 256, 0, stream>>>(
        xbN, lbN, pval, pidx);

    topk_merge_kernel<<<T_DIM / 256, 256, 0, stream>>>(pval, pidx, idx);

    gather_kernel<<<(C_DIM * T_DIM) / 256, 256, 0, stream>>>(P, pb, idx,
                                                             (float*)d_out);
}
